// CSON_41034117546379
// MI455X (gfx1250) — matrix operand order repaired and checked by reading, not compiled, not run on hardware
//
#include <hip/hip_runtime.h>
#include <hip/hip_bf16.h>
#include <math.h>

typedef __attribute__((ext_vector_type(16))) _Float16 v16h;
typedef __attribute__((ext_vector_type(8)))  _Float16 v8h;
typedef __attribute__((ext_vector_type(8)))  float    v8f;

#define NB   1024   // batch
#define TT   512    // eval points
#define CTXN 23
#define SS   24     // sequence (CLS + CTX)
#define DD   320
#define NL   5
#define NHD  8
#define DHD  40
#define FFN  1280
#define NCO  49     // NMODES+1

// f16 weight workspace layout (element offsets)
#define OFF_WIN   0
#define OFF_WOUT  (NL * 960 * DD)                 // 1,536,000
#define OFF_W1    (OFF_WOUT + NL * DD * DD)       // 2,048,000
#define OFF_W2    (OFF_W1   + NL * FFN * DD)      // 4,096,000
#define CVT_TOTAL (OFF_W2   + NL * DD * FFN)      // 6,144,000 halves = 12.3 MB

// LDS layout (bytes)
#define LDS_X    0                                 // 24*320 f32   = 30720
#define LDS_A16  30720                             // 32*320 f16   = 20480
#define LDS_BUF  51200                             // 24*960 f32 (92160) / 32*1280 f16 (81920)
#define LDS_SC   143360                            // 8*24*24 f32  = 18432
#define LDS_TOT  161792

__global__ void cvt_weights(const float* __restrict__ win, const float* __restrict__ wout,
                            const float* __restrict__ w1,  const float* __restrict__ w2,
                            _Float16* __restrict__ o) {
    size_t i = (size_t)blockIdx.x * blockDim.x + threadIdx.x;
    if (i >= (size_t)CVT_TOTAL) return;
    float v;
    if (i < OFF_WOUT)      v = win[i];
    else if (i < OFF_W1)   v = wout[i - OFF_WOUT];
    else if (i < OFF_W2)   v = w1[i - OFF_W1];
    else                   v = w2[i - OFF_W2];
    o[i] = (_Float16)v;
}

__device__ __forceinline__ float gelu_f(float v) {
    return 0.5f * v * (1.0f + erff(v * 0.70710678118654752f));
}

// A fragment (16x32 f16): lanes 0-15 row M=lane, K = k+0..7 (v0-3) and k+16..23 (v4-7)
//                         lanes 16-31 row M=lane-16, K = k+8..15 and k+24..31
__device__ __forceinline__ v16h load_a_frag(const _Float16* A, int lda, int m0, int k, int lane) {
    int row = m0 + (lane & 15);
    int off = (lane & 16) ? 8 : 0;
    const _Float16* p = A + row * lda + k + off;
    union { v16h v; v8h h[2]; } u;
    u.h[0] = *(const v8h*)(p);       // ds_load_b128
    u.h[1] = *(const v8h*)(p + 16);  // ds_load_b128
    return u.v;
}

// B fragment (32x16 f16) from row-major W[N][K] (computing x @ W.T):
// lane l<16: column n0+l, K = k+0..7 and k+16..23; lane l>=16: column n0+l-16, K = k+8..15 and k+24..31
__device__ __forceinline__ v16h load_b_frag(const _Float16* __restrict__ W, int ldw,
                                            int n0, int k, int lane) {
    int col = n0 + (lane & 15);
    int off = (lane & 16) ? 8 : 0;
    const _Float16* p = W + (size_t)col * ldw + k + off;
    union { v16h v; v8h h[2]; } u;
    u.h[0] = *(const v8h*)(p);
    u.h[1] = *(const v8h*)(p + 16);
    return u.v;  // 2x global_load_b128
}

// C = A(24x K, padded to 32) @ W.T (N x K).  Each wave owns strided N-tiles, keeps both
// M-tile accumulators live so each B fragment is reused twice (halves L2 weight traffic).
template <class F>
__device__ __forceinline__ void gemm_rows24(const _Float16* A, int lda,
                                            const _Float16* __restrict__ W,
                                            int N, int K, int wave, int lane, F&& emit) {
    const int mhi = (lane & 16) ? 8 : 0;
    const int nn  = lane & 15;
    for (int nt = wave; nt < (N >> 4); nt += 8) {
        const int n0 = nt << 4;
        v8f c0 = {}; v8f c1 = {};
        for (int k = 0; k < K; k += 32) {
            v16h b  = load_b_frag(W, K, n0, k, lane);
            v16h a0 = load_a_frag(A, lda, 0,  k, lane);
            v16h a1 = load_a_frag(A, lda, 16, k, lane);
            c0 = __builtin_amdgcn_wmma_f32_16x16x32_f16(false, a0, false, b, (short)0, c0, false, false);
            c1 = __builtin_amdgcn_wmma_f32_16x16x32_f16(false, a1, false, b, (short)0, c1, false, false);
        }
#pragma unroll
        for (int r = 0; r < 8; ++r) {
            emit(r + mhi, n0 + nn, c0[r]);          // rows 0..15: always valid
            int m1 = 16 + r + mhi;                   // rows 16..31: only <24 valid
            if (m1 < SS) emit(m1, n0 + nn, c1[r]);
        }
    }
}

__device__ __forceinline__ void layernorm_rows(const float* x, _Float16* out,
                                               const float* __restrict__ w,
                                               const float* __restrict__ b,
                                               int wave, int lane) {
    for (int r = wave; r < SS; r += 8) {
        float s = 0.f, s2 = 0.f;
        for (int i = lane; i < DD; i += 32) { float v = x[r * DD + i]; s += v; s2 += v * v; }
        for (int o = 16; o; o >>= 1) { s += __shfl_xor(s, o); s2 += __shfl_xor(s2, o); }
        float mean = s * (1.f / DD);
        float var  = s2 * (1.f / DD) - mean * mean;
        float rs   = rsqrtf(var + 1e-5f);
        for (int i = lane; i < DD; i += 32) {
            float v = (x[r * DD + i] - mean) * rs * w[i] + b[i];
            out[r * DD + i] = (_Float16)v;
        }
    }
}

__global__ __launch_bounds__(256)
void transformer_fwd(const float* __restrict__ k_norm, const float* __restrict__ ctx,
                     const float* __restrict__ W_e,    const float* __restrict__ b_e,
                     const float* __restrict__ cls,
                     const float* __restrict__ ln1_w,  const float* __restrict__ ln1_b,
                     const float* __restrict__ b_in,   const float* __restrict__ b_out,
                     const float* __restrict__ ln2_w,  const float* __restrict__ ln2_b,
                     const float* __restrict__ b1,     const float* __restrict__ b2,
                     const float* __restrict__ hln_w,  const float* __restrict__ hln_b,
                     const float* __restrict__ Wh1,    const float* __restrict__ bh1,
                     const float* __restrict__ Wh2,    const float* __restrict__ bh2,
                     const _Float16* __restrict__ w16, float* __restrict__ out) {
    extern __shared__ unsigned char smem[];
    float*    x_s   = (float*)(smem + LDS_X);     // 24x320 residual stream (f32)
    _Float16* a16   = (_Float16*)(smem + LDS_A16); // 32x320 WMMA A staging (f16)
    float*    buf   = (float*)(smem + LDS_BUF);   // 24x960 qkv f32
    _Float16* hid16 = (_Float16*)buf;              //  alias: 32x1280 FFN hidden f16
    float*    sc    = (float*)(smem + LDS_SC);    // 8x24x24 scores / head scratch

    const int tid  = threadIdx.x;
    const int wave = tid >> 5, lane = tid & 31;
    const int b    = blockIdx.x;

    // ---- embedding: row 0 = cls, rows 1..23 = ctx[t]*W_e + b_e ----
    for (int i = tid; i < SS * DD; i += 256) {
        int m = i / DD, d = i - m * DD;
        x_s[i] = (m == 0) ? cls[d] : (ctx[b * CTXN + (m - 1)] * W_e[d] + b_e[d]);
    }
    __syncthreads();

    for (int l = 0; l < NL; ++l) {
        const _Float16* Win  = w16 + OFF_WIN  + (size_t)l * 960 * DD;
        const _Float16* Wout = w16 + OFF_WOUT + (size_t)l * DD * DD;
        const _Float16* W1p  = w16 + OFF_W1   + (size_t)l * FFN * DD;
        const _Float16* W2p  = w16 + OFF_W2   + (size_t)l * DD * FFN;

        // ---- LN1 -> a16 ----
        layernorm_rows(x_s, a16, ln1_w + l * DD, ln1_b + l * DD, wave, lane);
        __syncthreads();

        // ---- QKV GEMM (24x320 @ 320x960) -> buf f32 ----
        const float* binL = b_in + l * 3 * DD;
        gemm_rows24(a16, DD, Win, 3 * DD, DD, wave, lane,
                    [&](int m, int n, float v) { buf[m * 960 + n] = v + binL[n]; });
        __syncthreads();

        // ---- attention scores (K=40 per head: VALU) ----
        const float scale = 0.15811388300841897f;  // 1/sqrt(40)
        for (int i = tid; i < NHD * SS * SS; i += 256) {
            int h = i / (SS * SS); int rem = i - h * SS * SS;
            int qi = rem / SS, kj = rem - qi * SS;
            const float* qp = buf + qi * 960 + h * DHD;
            const float* kp = buf + kj * 960 + DD + h * DHD;
            float s = 0.f;
            for (int d = 0; d < DHD; ++d) s += qp[d] * kp[d];
            sc[i] = s * scale;
        }
        __syncthreads();

        // ---- softmax per (head,row): 192 rows ----
        for (int i = tid; i < NHD * SS; i += 256) {
            float* row = sc + i * SS;
            float mx = row[0];
            for (int j = 1; j < SS; ++j) mx = fmaxf(mx, row[j]);
            float sum = 0.f;
            for (int j = 0; j < SS; ++j) { float e = __expf(row[j] - mx); row[j] = e; sum += e; }
            float inv = 1.f / sum;
            for (int j = 0; j < SS; ++j) row[j] *= inv;
        }
        __syncthreads();

        // ---- o = A @ V -> a16 (f16, ready as next WMMA A operand) ----
        for (int i = tid; i < SS * DD; i += 256) {
            int qi = i / DD, d = i - qi * DD;
            int h = d / DHD;
            const float* arow = sc + (h * SS + qi) * SS;
            float o = 0.f;
            for (int kj = 0; kj < SS; ++kj) o += arow[kj] * buf[kj * 960 + 2 * DD + d];
            a16[qi * DD + d] = (_Float16)o;
        }
        __syncthreads();

        // ---- out-proj GEMM + residual ----
        const float* boutL = b_out + l * DD;
        gemm_rows24(a16, DD, Wout, DD, DD, wave, lane,
                    [&](int m, int n, float v) { x_s[m * DD + n] += v + boutL[n]; });
        __syncthreads();

        // ---- LN2 -> a16 ----
        layernorm_rows(x_s, a16, ln2_w + l * DD, ln2_b + l * DD, wave, lane);
        __syncthreads();

        // ---- FFN1 GEMM + GELU -> hid16 ----
        const float* b1L = b1 + l * FFN;
        gemm_rows24(a16, DD, W1p, FFN, DD, wave, lane,
                    [&](int m, int n, float v) { hid16[m * FFN + n] = (_Float16)gelu_f(v + b1L[n]); });
        __syncthreads();

        // ---- FFN2 GEMM + residual ----
        const float* b2L = b2 + l * DD;
        gemm_rows24(hid16, FFN, W2p, DD, FFN, wave, lane,
                    [&](int m, int n, float v) { x_s[m * DD + n] += v + b2L[n]; });
        __syncthreads();
    }

    // ---- head: CLS LN -> MLP -> 49 Chebyshev coeffs (VALU, tiny) ----
    float* hbuf = sc;            // 320
    float* hid  = sc + DD;       // 320
    float* cbuf = sc + 2 * DD;   // 49
    if (wave == 0) {
        float s = 0.f, s2 = 0.f;
        for (int i = lane; i < DD; i += 32) { float v = x_s[i]; s += v; s2 += v * v; }
        for (int o = 16; o; o >>= 1) { s += __shfl_xor(s, o); s2 += __shfl_xor(s2, o); }
        float mean = s * (1.f / DD), var = s2 * (1.f / DD) - mean * mean;
        float rs = rsqrtf(var + 1e-5f);
        for (int i = lane; i < DD; i += 32)
            hbuf[i] = (x_s[i] - mean) * rs * hln_w[i] + hln_b[i];
    }
    __syncthreads();
    for (int j = tid; j < DD; j += 256) {
        const float* wr = Wh1 + (size_t)j * DD;
        float acc = bh1[j];
        for (int i = 0; i < DD; ++i) acc += hbuf[i] * wr[i];
        hid[j] = gelu_f(acc);
    }
    __syncthreads();
    if (tid < NCO) {
        const float* wr = Wh2 + (size_t)tid * DD;
        float acc = bh2[tid];
        for (int i = 0; i < DD; ++i) acc += hid[i] * wr[i];
        cbuf[tid] = acc;
    }
    __syncthreads();

    // ---- Chebyshev (Clenshaw-style recurrence) + sigmoid -> out (B,T) ----
    for (int t = tid; t < TT; t += 256) {
        float xx = k_norm[(size_t)b * TT + t];
        float tp = 1.f, tc = xx;
        float acc = cbuf[0] + cbuf[1] * xx;
        for (int n = 2; n < NCO; ++n) {
            float tn = 2.f * xx * tc - tp;
            acc += cbuf[n] * tn;
            tp = tc; tc = tn;
        }
        out[(size_t)b * TT + t] = 1.f / (1.f + __expf(-acc));
    }
}

extern "C" void kernel_launch(void* const* d_in, const int* in_sizes, int n_in,
                              void* d_out, int out_size, void* d_ws, size_t ws_size,
                              hipStream_t stream) {
    (void)in_sizes; (void)n_in; (void)out_size; (void)ws_size;
    const float* k_norm = (const float*)d_in[0];
    const float* ctx    = (const float*)d_in[1];
    const float* W_e    = (const float*)d_in[2];
    const float* b_e    = (const float*)d_in[3];
    const float* cls    = (const float*)d_in[4];
    const float* ln1_w  = (const float*)d_in[5];
    const float* ln1_b  = (const float*)d_in[6];
    const float* W_in   = (const float*)d_in[7];
    const float* b_in   = (const float*)d_in[8];
    const float* W_out  = (const float*)d_in[9];
    const float* b_out  = (const float*)d_in[10];
    const float* ln2_w  = (const float*)d_in[11];
    const float* ln2_b  = (const float*)d_in[12];
    const float* W1     = (const float*)d_in[13];
    const float* b1     = (const float*)d_in[14];
    const float* W2     = (const float*)d_in[15];
    const float* b2     = (const float*)d_in[16];
    const float* hln_w  = (const float*)d_in[17];
    const float* hln_b  = (const float*)d_in[18];
    const float* Wh1    = (const float*)d_in[19];
    const float* bh1    = (const float*)d_in[20];
    const float* Wh2    = (const float*)d_in[21];
    const float* bh2    = (const float*)d_in[22];
    float* out = (float*)d_out;
    _Float16* w16 = (_Float16*)d_ws;

    // 1) convert GEMM weights to f16 (already (N,K) row-major == WMMA B orientation)
    int nblk = (CVT_TOTAL + 255) / 256;
    cvt_weights<<<nblk, 256, 0, stream>>>(W_in, W_out, W1, W2, w16);

    // 2) fused transformer forward: 1 block per batch element, 8 wave32s, ~158 KB LDS
    transformer_fwd<<<NB, 256, LDS_TOT, stream>>>(
        k_norm, ctx, W_e, b_e, cls, ln1_w, ln1_b, b_in, b_out,
        ln2_w, ln2_b, b1, b2, hln_w, hln_b, Wh1, bh1, Wh2, bh2, w16, out);
}
